// DeformConv_2534030705071
// MI455X (gfx1250) — compile-verified, matrix operand order checked by reading, not run on hardware
//
#include <hip/hip_runtime.h>

// ---------------------------------------------------------------------------
// Deformable conv (3x3, pad 1, stride 1, dil 1, G=1) for gfx1250 (CDNA5).
// Fused bilinear-im2col + bf16 WMMA GEMM, f32 accumulate.
// v3: pair-gather — each bilinear row is one global_load_b64 (adjacent
// corners), with border cases folded into slot-remapped weights at metadata
// build time. Halves gather VMEM instruction count vs v2.
// ---------------------------------------------------------------------------

typedef __attribute__((ext_vector_type(16))) __bf16 v16bf;
typedef __attribute__((ext_vector_type(8)))  float  v8f;

#define NIMG   8
#define CIN    256
#define HH     64
#define WW     64
#define CO     256
#define KT     9           // 3x3 taps
#define HO     64
#define WO     64
#define HWSZ   (HH * WW)   // 4096
#define PIXIMG (HO * WO)   // 4096
#define TILE_P 128         // pixels per workgroup
#define CSTEP  32          // channels per reduce step (== WMMA K for bf16)
#define NITER  (KT * (CIN / CSTEP))  // 72

__device__ __forceinline__ __bf16 f2bf(float f) {
  // round-to-nearest-even fp32 -> bf16
  unsigned u = __builtin_bit_cast(unsigned, f);
  unsigned r = u + 0x7FFFu + ((u >> 16) & 1u);
  unsigned short h = (unsigned short)(r >> 16);
  return __builtin_bit_cast(__bf16, h);
}

__device__ __forceinline__ int iclamp(int v, int lo, int hi) {
  return v < lo ? lo : (v > hi ? hi : v);
}

// --------------------- Kernel 1: weight prepack ----------------------------
// apk row = ((kt*8 + cblk)*256 + co); entry = 32 bf16 (channels cblk*32..+31
// of weight row co at tap kt) == the per-lane K-contiguous A-fragment slice.
__global__ __launch_bounds__(256) void prepack_w(const float* __restrict__ wgt,
                                                 __bf16* __restrict__ apk) {
  const int idx = blockIdx.x * 256 + threadIdx.x;   // 72*256 rows
  const int kt  = idx >> 11;
  const int rem = idx & 2047;
  const int cb  = rem >> 8;
  const int co  = rem & 255;
  const float* src = wgt + (long)co * (CIN * KT) + (long)(cb * CSTEP) * KT + kt;
  v16bf v0, v1;
  #pragma unroll
  for (int j = 0; j < 16; ++j) v0[j] = f2bf(src[j * KT]);
  #pragma unroll
  for (int j = 0; j < 16; ++j) v1[j] = f2bf(src[(16 + j) * KT]);
  v16bf* dst = (v16bf*)(apk + (long)idx * CSTEP);
  dst[0] = v0;
  dst[1] = v1;
}

// --------------------- B-tile gather (pair loads + blend -> bf16 LDS) ------
__device__ __forceinline__ void gather_B(
    const float* __restrict__ x, long n_xbase,
    const float (*mw)[4], const unsigned* msp,
    __bf16 (*bdst)[CSTEP], int bp, int bc0, int it)
{
  const int kt = it >> 3;               // CIN/CSTEP == 8
  const int cc = (it & 7) * CSTEP;
  const int e  = bp * KT + kt;
  const float wl0 = mw[e][0], wh0 = mw[e][1];   // row y0: lo/hi slot weights
  const float wl1 = mw[e][2], wh1 = mw[e][3];   // row y1
  const unsigned sp = msp[e];
  const int b0 = (int)(sp & 0xFFFFu);           // cy0*W + cx0c
  const int b1 = (int)(sp >> 16);               // cy1*W + cx0c
  const float* xb = x + n_xbase + (long)(cc + bc0) * HWSZ;
  v16bf v;
  #pragma unroll
  for (int ci = 0; ci < 16; ++ci) {
    const float* xc = xb + (long)ci * HWSZ;
    float p0[2], p1[2];
    __builtin_memcpy(p0, xc + b0, 8);   // global_load_b64 (align 4 ok)
    __builtin_memcpy(p1, xc + b1, 8);
    v[ci] = f2bf(wl0 * p0[0] + wh0 * p0[1] + wl1 * p1[0] + wh1 * p1[1]);
  }
  *(v16bf*)&bdst[bp][bc0] = v;          // 32B vector store -> ds_store_b128 x2
}

// --------------------- Kernel 2: fused deform-conv GEMM --------------------
__global__ __launch_bounds__(256) void deform_conv_wmma(
    const float*  __restrict__ x,     // [N, C, H, W]
    const float*  __restrict__ off,   // [N, 2*KT, Ho, Wo]
    const __bf16* __restrict__ apk,   // prepacked weights (kernel 1)
    float*        __restrict__ out)   // [N, Co, Ho, Wo]
{
  // Per-(pixel,tap) metadata: 4 slot-remapped bilinear weights + packed u32
  // of the two clamped row base offsets (16-bit y*W + clamp(x0,0,W-2)).
  __shared__ __align__(16) float     mw[TILE_P * KT][4];
  __shared__ __align__(16) unsigned  msp[TILE_P * KT];
  // Double-buffered im2col tile, K-contiguous per pixel row.
  __shared__ __align__(64) __bf16    Bbuf[2][TILE_P][CSTEP];

  const int tid       = threadIdx.x;
  const int tile_base = blockIdx.x * TILE_P;      // flat pixel base
  const int n         = tile_base >> 12;          // tile never crosses an image
  const int rem_base  = tile_base & 4095;
  const long n_xbase  = (long)n * CIN * HWSZ;

  // ---- Stage 0: bilinear metadata with border-slot folding ----------------
  for (int e = tid; e < TILE_P * KT; e += blockDim.x) {
    const int p  = e / KT;
    const int kt = e - p * KT;
    const int gp = rem_base + p;
    const int ho = gp >> 6, wo = gp & 63;
    const int kh = kt / 3,  kw = kt - kh * 3;
    const float oy = off[(((n * 18) + 2 * kt) * HO + ho) * WO + wo];
    const float ox = off[(((n * 18) + 2 * kt + 1) * HO + ho) * WO + wo];
    const float py = (float)(ho - 1 + kh) + oy;   // pad=1, stride=1, dil=1
    const float px = (float)(wo - 1 + kw) + ox;
    const float fy = floorf(py), fx = floorf(px);
    const int iy0 = (int)fy, ix0 = (int)fx;
    const float wy = py - fy, wx = px - fx;
    // x-direction weights with zero-pad validity
    const float vx0 = (ix0     >= 0 && ix0     < WW) ? 1.f : 0.f;
    const float vx1 = (ix0 + 1 >= 0 && ix0 + 1 < WW) ? 1.f : 0.f;
    float ul = (1.f - wx) * vx0;       // weight for pair slot 0 (x[cx0c])
    float uh = wx * vx1;               // weight for pair slot 1 (x[cx0c+1])
    if (ix0 >= WW - 1) { uh = ul; ul = 0.f; }  // valid corner sits in hi slot
    if (ix0 < 0)       { ul = uh; uh = 0.f; }  // valid corner sits in lo slot
    // y-direction weights with validity
    const float ry0 = (1.f - wy) * ((iy0     >= 0 && iy0     < HH) ? 1.f : 0.f);
    const float ry1 = wy         * ((iy0 + 1 >= 0 && iy0 + 1 < HH) ? 1.f : 0.f);
    mw[e][0] = ry0 * ul;
    mw[e][1] = ry0 * uh;
    mw[e][2] = ry1 * ul;
    mw[e][3] = ry1 * uh;
    const int cx0c = iclamp(ix0,     0, WW - 2);   // pair base (always safe +1)
    const int cy0  = iclamp(iy0,     0, HH - 1);
    const int cy1  = iclamp(iy0 + 1, 0, HH - 1);
    msp[e] = (unsigned)((cy0 * WW + cx0c) | ((cy1 * WW + cx0c) << 16));
  }
  __syncthreads();

  // ---- Wave tiling: 8 waves as 4(M) x 2(N); each wave owns 64x64 ----------
  const int lane  = tid & 31;
  const int wid   = tid >> 5;
  const int wm    = (wid >> 1) * 64;      // Co base of this wave
  const int wn    = (wid & 1)  * 64;      // pixel base of this wave
  const int frow  = lane & 15;            // row/col within a 16-wide fragment
  const int fksel = (lane >> 4) * 16;     // K half: lanes 0-15 -> K0..15

  v8f acc[4][4];
  #pragma unroll
  for (int i = 0; i < 4; ++i)
    #pragma unroll
    for (int j = 0; j < 4; ++j)
      acc[i][j] = (v8f){0.f, 0.f, 0.f, 0.f, 0.f, 0.f, 0.f, 0.f};

  // B-build mapping: 2 threads per pixel, 16 channels each.
  const int bp  = tid >> 1;
  const int bc0 = (tid & 1) * 16;

  // ---- Pipelined main reduction (one barrier/iter, gather ahead by 1) -----
  gather_B(x, n_xbase, mw, msp, Bbuf[0], bp, bc0, 0);
  __syncthreads();

  for (int it = 0; it < NITER; ++it) {
    if (it + 1 < NITER)
      gather_B(x, n_xbase, mw, msp, Bbuf[(it + 1) & 1], bp, bc0, it + 1);

    // A fragments: direct 32B global loads from prepacked bf16 weights.
    const __bf16* arow = apk + (long)it * (CO * CSTEP);
    __builtin_prefetch(arow + CO * CSTEP, 0, 0);   // next iteration's A chunk

    v16bf afrag[4], bfrag[4];
    #pragma unroll
    for (int i = 0; i < 4; ++i)
      afrag[i] = *(const v16bf*)(arow + (wm + i * 16 + frow) * CSTEP + fksel);
    #pragma unroll
    for (int j = 0; j < 4; ++j)
      bfrag[j] = *(const v16bf*)&Bbuf[it & 1][wn + j * 16 + frow][fksel];

    #pragma unroll
    for (int i = 0; i < 4; ++i)
      #pragma unroll
      for (int j = 0; j < 4; ++j)
        acc[i][j] = __builtin_amdgcn_wmma_f32_16x16x32_bf16(
            false, afrag[i], false, bfrag[j],
            (short)0, acc[i][j], false, false);

    __syncthreads();   // releases Bbuf[it&1] for overwrite next iteration
  }

  // ---- Epilogue: f32 accumulators -> out[n][co][ho][wo] -------------------
  // C/D layout: VGPR r -> lanes 0-15: M=r, N=lane; lanes 16-31: M=8+r.
  float* obase = out + (long)n * CO * PIXIMG + rem_base;
  const int mrow = (lane >> 4) * 8;
  const int ncol = lane & 15;
  #pragma unroll
  for (int i = 0; i < 4; ++i)
    #pragma unroll
    for (int j = 0; j < 4; ++j)
      #pragma unroll
      for (int r = 0; r < 8; ++r) {
        const int co = wm + i * 16 + mrow + r;
        const int pl = wn + j * 16 + ncol;
        obase[(long)co * PIXIMG + pl] = acc[i][j][r];
      }
}

extern "C" void kernel_launch(void* const* d_in, const int* in_sizes, int n_in,
                              void* d_out, int out_size, void* d_ws, size_t ws_size,
                              hipStream_t stream) {
  (void)in_sizes; (void)n_in; (void)out_size; (void)ws_size;
  const float* x   = (const float*)d_in[0];   // [8,256,64,64]
  const float* off = (const float*)d_in[1];   // [8,18,64,64]
  const float* wgt = (const float*)d_in[2];   // [256,256,3,3]
  float* out  = (float*)d_out;                // [8,256,64,64]
  __bf16* apk = (__bf16*)d_ws;                // 9*8*256*32 bf16 = 1.18 MB

  // Kernel 1: prepack weights into WMMA A-fragment layout.
  prepack_w<<<dim3(NITER), dim3(256), 0, stream>>>(wgt, apk);

  // Kernel 2: fused bilinear-im2col + WMMA GEMM, one WG per 128-pixel tile.
  const int n_tiles = (NIMG * PIXIMG) / TILE_P;  // 256 workgroups
  deform_conv_wmma<<<dim3(n_tiles), dim3(256), 0, stream>>>(x, off, apk, out);
}
